// AttentionPairBias_18124761989723
// MI455X (gfx1250) — matrix operand order repaired and checked by reading, not compiled, not run on hardware
//
#include <hip/hip_runtime.h>

typedef __bf16 bf16_t;
typedef __attribute__((ext_vector_type(16))) __bf16 v16bf;
typedef __attribute__((ext_vector_type(8)))  float  v8f;
typedef unsigned long long u64_t;

#define NTOK 1024
#define CA   768
#define CS   384
#define CZ   128
#define NH   16
#define HD   48
#define HDP  64   // head dim padded to WMMA K multiple

__device__ __forceinline__ v8f wmma_bf16(v16bf a, v16bf b, v8f c) {
  return __builtin_amdgcn_wmma_f32_16x16x32_bf16(false, a, false, b, (short)0, c, false, false);
}

__device__ __forceinline__ unsigned pack2(float a, float b) {
  union { bf16_t h[2]; unsigned u; } p;
  p.h[0] = (bf16_t)a; p.h[1] = (bf16_t)b;
  return p.u;
}

// ---- fragment loads: each half of a fragment is 16 contiguous bytes -> 2x b128 ----
__device__ __forceinline__ v16bf load_a_frag_s(const bf16_t* As, int str) {
  int lane = threadIdx.x & 31, m = lane & 15, hl = lane >> 4;
  union { uint4 q[2]; v16bf v; } f;
  f.q[0] = *(const uint4*)(As + m * str + hl * 8);
  f.q[1] = *(const uint4*)(As + m * str + 16 + hl * 8);
  return f.v;
}
__device__ __forceinline__ v16bf load_b_frag_s(const bf16_t* BsN, int str) {
  int lane = threadIdx.x & 31, n = lane & 15, hl = lane >> 4;
  union { uint4 q[2]; v16bf v; } f;
  f.q[0] = *(const uint4*)(BsN + n * str + hl * 8);
  f.q[1] = *(const uint4*)(BsN + n * str + 16 + hl * 8);
  return f.v;
}
__device__ __forceinline__ v16bf load_a_frag(const bf16_t* As) { return load_a_frag_s(As, 32); }
__device__ __forceinline__ v16bf load_b_frag(const bf16_t* BsN) { return load_b_frag_s(BsN, 32); }

// ---- LDS staging: straight-line 8-byte block copies, no exec-masked loops ----
__device__ __forceinline__ void stage_a16x32(bf16_t* As, const bf16_t* A, int lda, int tid) {
  if (tid < 64) {
    int r = tid >> 2, c = (tid & 3) * 8;
    *(u64_t*)(As + r * 32 + c) = *(const u64_t*)(A + (size_t)r * lda + c);
  }
}
// BT row-major [N][K]; copy 64 rows x 16 of 32 K (4 of every 8 K); 128 threads -> exactly 2 fixed chunks each
__device__ __forceinline__ void stage_bt64x32(bf16_t* Bs, const bf16_t* BT, int ldk, int tid) {
  {
    int r = tid >> 2, c = (tid & 3) * 8;
    *(u64_t*)(Bs + r * 32 + c) = *(const u64_t*)(BT + (size_t)r * ldk + c);
  }
  {
    int i = tid + 128;
    int r = i >> 2, c = (i & 3) * 8;
    *(u64_t*)(Bs + r * 32 + c) = *(const u64_t*)(BT + (size_t)r * ldk + c);
  }
}

// ---------------- weight convert + transpose: y[n][k] = bf16(x[k][n]), N = 768 ----------------
__global__ void cvt_t_kernel(const float* __restrict__ x, bf16_t* __restrict__ y, int K) {
  int k = blockIdx.x;
  int n = blockIdx.y * 256 + threadIdx.x;
  y[(size_t)n * K + k] = (bf16_t)x[(size_t)k * CA + n];
}
__global__ void zero4_kernel(uint4* __restrict__ y, int n4) {
  int i = blockIdx.x * 256 + threadIdx.x;
  if (i < n4) y[i] = make_uint4(0, 0, 0, 0);
}

// ---------------- row LayerNorms ----------------
__global__ __launch_bounds__(256) void ln_a_kernel(const float* __restrict__ a, float* __restrict__ an) {
  __shared__ float rs[256], rq[256];
  int row = blockIdx.x, tid = threadIdx.x;
  const float* x = a + (size_t)row * CA;
  float s = 0.f, q = 0.f, v0[3];
#pragma unroll
  for (int i = 0; i < 3; ++i) { float t = x[tid + i * 256]; v0[i] = t; s += t; q += t * t; }
  rs[tid] = s; rq[tid] = q; __syncthreads();
  for (int st = 128; st > 0; st >>= 1) {
    if (tid < st) { rs[tid] += rs[tid + st]; rq[tid] += rq[tid + st]; }
    __syncthreads();
  }
  float mean = rs[0] / (float)CA;
  float var  = rq[0] / (float)CA - mean * mean;
  float inv  = rsqrtf(var + 1e-5f);
#pragma unroll
  for (int i = 0; i < 3; ++i) an[(size_t)row * CA + tid + i * 256] = (v0[i] - mean) * inv;
}

__global__ __launch_bounds__(128) void ln_s_kernel(const float* __restrict__ sIn, const float* __restrict__ s_ln_w,
                                                   bf16_t* __restrict__ sn, bf16_t* __restrict__ sbf) {
  __shared__ float rs[128], rq[128];
  int row = blockIdx.x, tid = threadIdx.x;
  const float* x = sIn + (size_t)row * CS;
  float s = 0.f, q = 0.f, v0[3];
#pragma unroll
  for (int i = 0; i < 3; ++i) { float t = x[tid + i * 128]; v0[i] = t; s += t; q += t * t; }
  rs[tid] = s; rq[tid] = q; __syncthreads();
  for (int st = 64; st > 0; st >>= 1) {
    if (tid < st) { rs[tid] += rs[tid + st]; rq[tid] += rq[tid + st]; }
    __syncthreads();
  }
  float mean = rs[0] / (float)CS;
  float var  = rq[0] / (float)CS - mean * mean;
  float inv  = rsqrtf(var + 1e-5f);
#pragma unroll
  for (int i = 0; i < 3; ++i) {
    int c = tid + i * 128;
    sn[(size_t)row * CS + c]  = (bf16_t)((v0[i] - mean) * inv * s_ln_w[c]);
    sbf[(size_t)row * CS + c] = (bf16_t)v0[i];
  }
}

// ---------------- adaLN fused dual-GEMM: a2 = sigmoid(sn@gate_w+gb)*a_n + sn@skip_w ----------------
__global__ __launch_bounds__(128) void adaln_gemm_kernel(const bf16_t* __restrict__ sn, const bf16_t* __restrict__ gwT,
                                                         const bf16_t* __restrict__ swT, const float* __restrict__ gate_b,
                                                         const float* __restrict__ an, bf16_t* __restrict__ a2) {
  __shared__ __align__(16) bf16_t As[16 * 32];
  __shared__ __align__(16) bf16_t Bs[2][64 * 32];
  int tid = threadIdx.x, wave = tid >> 5, lane = tid & 31;
  int m0 = blockIdx.x * 16, n0 = blockIdx.y * 64;
  v8f accg = {}, accs = {};
  for (int k0 = 0; k0 < CS; k0 += 32) {
    stage_a16x32(As, sn + (size_t)m0 * CS + k0, CS, tid);
    stage_bt64x32(Bs[0], gwT + (size_t)n0 * CS + k0, CS, tid);
    stage_bt64x32(Bs[1], swT + (size_t)n0 * CS + k0, CS, tid);
    __syncthreads();
    v16bf af = load_a_frag(As);
    accg = wmma_bf16(af, load_b_frag(&Bs[0][wave * 512]), accg);
    accs = wmma_bf16(af, load_b_frag(&Bs[1][wave * 512]), accs);
    __syncthreads();
  }
  int n = n0 + wave * 16 + (lane & 15), hl = lane >> 4;
#pragma unroll
  for (int r = 0; r < 8; ++r) {
    int m = m0 + hl * 8 + r;
    float gate = 1.f / (1.f + __expf(-(accg[r] + gate_b[n])));
    a2[(size_t)m * CA + n] = (bf16_t)(gate * an[(size_t)m * CA + n] + accs[r]);
  }
}

// ---------------- QKVG quad-GEMM ----------------
__global__ __launch_bounds__(128) void qkvg_kernel(const bf16_t* __restrict__ a2, const bf16_t* __restrict__ wqT,
                                                   const bf16_t* __restrict__ wkT, const bf16_t* __restrict__ wvT,
                                                   const bf16_t* __restrict__ wgT, const float* __restrict__ bq,
                                                   bf16_t* __restrict__ qpad, bf16_t* __restrict__ kpad,
                                                   bf16_t* __restrict__ vT, float* __restrict__ gT) {
  __shared__ __align__(16) bf16_t As[16 * 32];
  __shared__ __align__(16) bf16_t Bs[4][64 * 32];
  int tid = threadIdx.x, wave = tid >> 5, lane = tid & 31;
  int m0 = blockIdx.x * 16, n0 = blockIdx.y * 64;
  v8f aq = {}, ak = {}, av = {}, ag = {};
  for (int k0 = 0; k0 < CA; k0 += 32) {
    stage_a16x32(As, a2 + (size_t)m0 * CA + k0, CA, tid);
    stage_bt64x32(Bs[0], wqT + (size_t)n0 * CA + k0, CA, tid);
    stage_bt64x32(Bs[1], wkT + (size_t)n0 * CA + k0, CA, tid);
    stage_bt64x32(Bs[2], wvT + (size_t)n0 * CA + k0, CA, tid);
    stage_bt64x32(Bs[3], wgT + (size_t)n0 * CA + k0, CA, tid);
    __syncthreads();
    v16bf af = load_a_frag(As);
    aq = wmma_bf16(af, load_b_frag(&Bs[0][wave * 512]), aq);
    ak = wmma_bf16(af, load_b_frag(&Bs[1][wave * 512]), ak);
    av = wmma_bf16(af, load_b_frag(&Bs[2][wave * 512]), av);
    ag = wmma_bf16(af, load_b_frag(&Bs[3][wave * 512]), ag);
    __syncthreads();
  }
  int c = n0 + wave * 16 + (lane & 15), hl = lane >> 4;
  int h = c / HD, d = c - h * HD;
  const float scale = 0.14433756729740643f; // 1/sqrt(48), folded into q
#pragma unroll
  for (int r = 0; r < 8; ++r) {
    int m = m0 + hl * 8 + r;
    qpad[((size_t)h * NTOK + m) * HDP + d] = (bf16_t)((aq[r] + bq[c]) * scale);
    kpad[((size_t)h * NTOK + m) * HDP + d] = (bf16_t)ak[r];
    vT[((size_t)c) * NTOK + m] = (bf16_t)av[r];          // [h*48+d][tok]
    gT[((size_t)c) * NTOK + m] = 1.f / (1.f + __expf(-ag[r]));
  }
}

// ---------------- pair bias: LN(z) @ wz as a WMMA GEMM, one pass over z ----------------
__global__ __launch_bounds__(256) void pair_bias_kernel(const float* __restrict__ z, const float* __restrict__ zw,
                                                        const float* __restrict__ zb, const float* __restrict__ wz,
                                                        float* __restrict__ bias) {
  __shared__ __align__(16) bf16_t Az[32 * 128];   // normalized z, 32 positions x 128 ch
  __shared__ __align__(16) bf16_t WzT[16 * 128];  // wz transposed to [h][c]
  __shared__ float part[8 * 256];                 // per-wave partial 16x16 tiles
  int tid = threadIdx.x;
#pragma unroll
  for (int jj = 0; jj < 8; ++jj) {
    int i = tid + jj * 256;
    int h = i >> 7, c = i & 127;
    WzT[i] = (bf16_t)wz[c * NH + h];
  }
  int pl = tid >> 3, oct = tid & 7, c0 = oct * 16;
  size_t pos = (size_t)blockIdx.x * 32 + pl;
  const float* zp = z + pos * CZ + c0;
  __builtin_prefetch(zp + 32 * CZ, 0, 1);
  union { float4 f4[4]; float f[16]; } X, W, Bb;
#pragma unroll
  for (int j = 0; j < 4; ++j) {
    X.f4[j]  = *(const float4*)(zp + j * 4);
    W.f4[j]  = *(const float4*)(zw + c0 + j * 4);
    Bb.f4[j] = *(const float4*)(zb + c0 + j * 4);
  }
  float s = 0.f, q = 0.f;
#pragma unroll
  for (int j = 0; j < 16; ++j) { s += X.f[j]; q += X.f[j] * X.f[j]; }
#pragma unroll
  for (int off = 4; off >= 1; off >>= 1) { s += __shfl_xor(s, off, 32); q += __shfl_xor(q, off, 32); }
  float mean = s * (1.f / 128.f);
  float var  = q * (1.f / 128.f) - mean * mean;
  float inv  = rsqrtf(var + 1e-5f);
#pragma unroll
  for (int j = 0; j < 8; ++j) {
    float y0 = (X.f[2 * j]     - mean) * inv * W.f[2 * j]     + Bb.f[2 * j];
    float y1 = (X.f[2 * j + 1] - mean) * inv * W.f[2 * j + 1] + Bb.f[2 * j + 1];
    *(unsigned*)(Az + pl * 128 + c0 + 2 * j) = pack2(y0, y1);
  }
  __syncthreads();
  int wave = tid >> 5, lane = tid & 31;
  int tile = wave >> 2, ks = wave & 3;          // 2 M-tiles x 4 K-steps across 8 waves
  v8f acc = {};
  {
    v16bf af  = load_a_frag_s(Az + tile * 16 * 128 + ks * 32, 128);
    v16bf bfr = load_b_frag_s(WzT + ks * 32, 128);
    acc = wmma_bf16(af, bfr, acc);
  }
  int hl = lane >> 4, n = lane & 15;
#pragma unroll
  for (int r = 0; r < 8; ++r) part[wave * 256 + (hl * 8 + r) * 16 + n] = acc[r];
  __syncthreads();
#pragma unroll
  for (int tl = 0; tl < 2; ++tl) {
    int elem = tid;
    float sum = part[(tl * 4 + 0) * 256 + elem] + part[(tl * 4 + 1) * 256 + elem] +
                part[(tl * 4 + 2) * 256 + elem] + part[(tl * 4 + 3) * 256 + elem];
    int m = elem >> 4, hcol = elem & 15;
    size_t pp = (size_t)blockIdx.x * 32 + (size_t)tl * 16 + m;
    size_t i = pp >> 10, j = pp & 1023;
    bias[((size_t)hcol * NTOK + i) * NTOK + j] = sum;  // [H][N][N]
  }
}

// ---------------- QK^T + bias + softmax (in place over bias buffer) ----------------
__global__ __launch_bounds__(256) void attn_softmax_kernel(const bf16_t* __restrict__ qpad,
                                                           const bf16_t* __restrict__ kpad,
                                                           float* __restrict__ bias) {
  __shared__ __align__(16) bf16_t Qs[16 * HDP];
  __shared__ __align__(16) float lg[16 * NTOK];
  __shared__ float red[256];
  int tid = threadIdx.x, wave = tid >> 5, lane = tid & 31;
  int h = blockIdx.y, q0 = blockIdx.x * 16;
  {  // exactly one 8-byte chunk per thread
    int r = tid >> 4, c = (tid & 15) * 4;
    *(u64_t*)(Qs + r * HDP + c) = *(const u64_t*)(qpad + ((size_t)h * NTOK + q0 + r) * HDP + c);
  }
  __syncthreads();
  int m = lane & 15, hl = lane >> 4;
  for (int pass = 0; pass < 8; ++pass) {
    int kb = (pass * 8 + wave) * 16;
    v8f acc = {};
#pragma unroll
    for (int kk = 0; kk < HDP; kk += 32) {
      union { uint4 q[2]; v16bf v; } fa, fb;
      fa.q[0] = *(const uint4*)(Qs + m * HDP + kk + hl * 8);
      fa.q[1] = *(const uint4*)(Qs + m * HDP + kk + 16 + hl * 8);
      const bf16_t* kr = kpad + ((size_t)h * NTOK + kb + m) * HDP + kk;
      fb.q[0] = *(const uint4*)(kr + hl * 8);
      fb.q[1] = *(const uint4*)(kr + 16 + hl * 8);
      acc = wmma_bf16(fa.v, fb.v, acc);
    }
#pragma unroll
    for (int r = 0; r < 8; ++r) lg[(hl * 8 + r) * NTOK + kb + m] = acc[r];
  }
  __syncthreads();
  // softmax: 16 rows x 16 chunks of 64 cols (float4 throughout), bias added here
  int row = tid & 15, ch = tid >> 4;
  float4* lg4 = (float4*)(lg + row * NTOK + ch * 64);
  const float4* bs4 = (const float4*)(bias + ((size_t)h * NTOK + q0 + row) * NTOK + ch * 64);
  float mx = -3.0e38f;
  for (int c = 0; c < 16; ++c) {
    float4 l4 = lg4[c], b4 = bs4[c];
    l4.x += b4.x; l4.y += b4.y; l4.z += b4.z; l4.w += b4.w;
    lg4[c] = l4;
    mx = fmaxf(mx, fmaxf(fmaxf(l4.x, l4.y), fmaxf(l4.z, l4.w)));
  }
  red[ch * 16 + row] = mx; __syncthreads();
  for (int st = 8; st > 0; st >>= 1) {
    if (ch < st) red[ch * 16 + row] = fmaxf(red[ch * 16 + row], red[(ch + st) * 16 + row]);
    __syncthreads();
  }
  float rmax = red[row]; __syncthreads();
  float psum = 0.f;
  for (int c = 0; c < 16; ++c) {
    float4 l4 = lg4[c];
    l4.x = __expf(l4.x - rmax); l4.y = __expf(l4.y - rmax);
    l4.z = __expf(l4.z - rmax); l4.w = __expf(l4.w - rmax);
    lg4[c] = l4;
    psum += l4.x + l4.y + l4.z + l4.w;
  }
  red[ch * 16 + row] = psum; __syncthreads();
  for (int st = 8; st > 0; st >>= 1) {
    if (ch < st) red[ch * 16 + row] += red[(ch + st) * 16 + row];
    __syncthreads();
  }
  float inv = 1.f / red[row];
  float4* out4 = (float4*)(bias + ((size_t)h * NTOK + q0 + row) * NTOK + ch * 64);
  for (int c = 0; c < 16; ++c) {
    float4 l4 = lg4[c];
    l4.x *= inv; l4.y *= inv; l4.z *= inv; l4.w *= inv;
    out4[c] = l4;
  }
}

// ---------------- o = P @ V fused with sigmoid gate ----------------
__global__ __launch_bounds__(128) void pv_kernel(const float* __restrict__ p, const bf16_t* __restrict__ vT,
                                                 const float* __restrict__ gT, bf16_t* __restrict__ og) {
  int tid = threadIdx.x, wave = tid >> 5, lane = tid & 31;
  int h = blockIdx.y;
  int q0 = blockIdx.x * 64 + wave * 16;
  int m = lane & 15, hl = lane >> 4;
  v8f a0 = {}, a1 = {}, a2 = {};
  for (int k0 = 0; k0 < NTOK; k0 += 32) {
    union { uint4 q[2]; v16bf v; } fa, f0, f1, f2;
    {
      const float* pp = p + ((size_t)h * NTOK + q0 + m) * NTOK + k0;
      float4 x0 = *(const float4*)(pp + hl * 8);
      float4 x1 = *(const float4*)(pp + hl * 8 + 4);
      float4 x2 = *(const float4*)(pp + 16 + hl * 8);
      float4 x3 = *(const float4*)(pp + 16 + hl * 8 + 4);
      fa.q[0] = make_uint4(pack2(x0.x, x0.y), pack2(x0.z, x0.w), pack2(x1.x, x1.y), pack2(x1.z, x1.w));
      fa.q[1] = make_uint4(pack2(x2.x, x2.y), pack2(x2.z, x2.w), pack2(x3.x, x3.y), pack2(x3.z, x3.w));
    }
    int koff = k0 + hl * 8;
    const bf16_t* v0r = vT + ((size_t)h * HD + 0  + m) * NTOK + koff;
    const bf16_t* v1r = vT + ((size_t)h * HD + 16 + m) * NTOK + koff;
    const bf16_t* v2r = vT + ((size_t)h * HD + 32 + m) * NTOK + koff;
    f0.q[0] = *(const uint4*)(v0r); f0.q[1] = *(const uint4*)(v0r + 16);
    f1.q[0] = *(const uint4*)(v1r); f1.q[1] = *(const uint4*)(v1r + 16);
    f2.q[0] = *(const uint4*)(v2r); f2.q[1] = *(const uint4*)(v2r + 16);
    a0 = wmma_bf16(fa.v, f0.v, a0);
    a1 = wmma_bf16(fa.v, f1.v, a1);
    a2 = wmma_bf16(fa.v, f2.v, a2);
  }
#pragma unroll
  for (int t = 0; t < 3; ++t) {
    int c = h * HD + t * 16 + m;
    const float* gr = gT + (size_t)c * NTOK + q0 + hl * 8;
    float4 ga = *(const float4*)(gr), gb = *(const float4*)(gr + 4);
    float gg[8] = { ga.x, ga.y, ga.z, ga.w, gb.x, gb.y, gb.z, gb.w };
    v8f* acc = (t == 0) ? &a0 : (t == 1) ? &a1 : &a2;
#pragma unroll
    for (int r = 0; r < 8; ++r) {
      int tok = q0 + hl * 8 + r;
      og[(size_t)tok * CA + c] = (bf16_t)(gg[r] * (*acc)[r]);
    }
  }
}

// ---------------- final: out = sigmoid(s@last_w + last_b) * (og@wo + bo) ----------------
__global__ __launch_bounds__(128) void out_gemm_kernel(const bf16_t* __restrict__ og, const bf16_t* __restrict__ woT,
                                                       const float* __restrict__ bo, const bf16_t* __restrict__ sbf,
                                                       const bf16_t* __restrict__ lwT, const float* __restrict__ last_b,
                                                       float* __restrict__ out) {
  __shared__ __align__(16) bf16_t As[16 * 32];
  __shared__ __align__(16) bf16_t Bs[64 * 32];
  int tid = threadIdx.x, wave = tid >> 5, lane = tid & 31;
  int m0 = blockIdx.x * 16, n0 = blockIdx.y * 64;
  v8f acc1 = {}, acc2 = {};
  for (int k0 = 0; k0 < CA; k0 += 32) {
    stage_a16x32(As, og + (size_t)m0 * CA + k0, CA, tid);
    stage_bt64x32(Bs, woT + (size_t)n0 * CA + k0, CA, tid);
    __syncthreads();
    acc1 = wmma_bf16(load_a_frag(As), load_b_frag(Bs + wave * 512), acc1);
    __syncthreads();
  }
  for (int k0 = 0; k0 < CS; k0 += 32) {
    stage_a16x32(As, sbf + (size_t)m0 * CS + k0, CS, tid);
    stage_bt64x32(Bs, lwT + (size_t)n0 * CS + k0, CS, tid);
    __syncthreads();
    acc2 = wmma_bf16(load_a_frag(As), load_b_frag(Bs + wave * 512), acc2);
    __syncthreads();
  }
  int n = n0 + wave * 16 + (lane & 15), hl = lane >> 4;
#pragma unroll
  for (int r = 0; r < 8; ++r) {
    int m = m0 + hl * 8 + r;
    float gate = 1.f / (1.f + __expf(-(acc2[r] + last_b[n])));
    out[(size_t)m * CA + n] = gate * (acc1[r] + bo[n]);
  }
}

extern "C" void kernel_launch(void* const* d_in, const int* in_sizes, int n_in,
                              void* d_out, int out_size, void* d_ws, size_t ws_size,
                              hipStream_t stream) {
  const float* a      = (const float*)d_in[0];
  const float* s      = (const float*)d_in[1];
  const float* z      = (const float*)d_in[2];
  const float* s_ln_w = (const float*)d_in[3];
  const float* gate_w = (const float*)d_in[4];
  const float* gate_b = (const float*)d_in[5];
  const float* skip_w = (const float*)d_in[6];
  const float* wq     = (const float*)d_in[7];
  const float* bq     = (const float*)d_in[8];
  const float* wk     = (const float*)d_in[9];
  const float* wv     = (const float*)d_in[10];
  const float* wg     = (const float*)d_in[11];
  const float* wo     = (const float*)d_in[12];
  const float* bo     = (const float*)d_in[13];
  const float* z_ln_w = (const float*)d_in[14];
  const float* z_ln_b = (const float*)d_in[15];
  const float* wz     = (const float*)d_in[16];
  const float* last_w = (const float*)d_in[17];
  const float* last_b = (const float*)d_in[18];
  float* out = (float*)d_out;

  char* w = (char*)d_ws;
  size_t off = 0;
  auto alloc = [&](size_t bytes) -> void* {
    void* p = w + off;
    off = (off + bytes + 255) & ~(size_t)255;
    return p;
  };
  const int W_SA = CS * CA;
  const int W_AA = CA * CA;
  bf16_t* gwT  = (bf16_t*)alloc((size_t)W_SA * 2);
  bf16_t* swT  = (bf16_t*)alloc((size_t)W_SA * 2);
  bf16_t* wqT  = (bf16_t*)alloc((size_t)W_AA * 2);
  bf16_t* wkT  = (bf16_t*)alloc((size_t)W_AA * 2);
  bf16_t* wvT  = (bf16_t*)alloc((size_t)W_AA * 2);
  bf16_t* wgT  = (bf16_t*)alloc((size_t)W_AA * 2);
  bf16_t* woT  = (bf16_t*)alloc((size_t)W_AA * 2);
  bf16_t* lwT  = (bf16_t*)alloc((size_t)W_SA * 2);
  bf16_t* snb  = (bf16_t*)alloc((size_t)NTOK * CS * 2);
  bf16_t* sbb  = (bf16_t*)alloc((size_t)NTOK * CS * 2);
  float*  anf  = (float*) alloc((size_t)NTOK * CA * 4);
  bf16_t* a2b  = (bf16_t*)alloc((size_t)NTOK * CA * 2);
  bf16_t* qpad = (bf16_t*)alloc((size_t)NH * NTOK * HDP * 2);
  bf16_t* kpad = (bf16_t*)alloc((size_t)NH * NTOK * HDP * 2);
  bf16_t* vT   = (bf16_t*)alloc((size_t)NTOK * CA * 2);   // [h*48+d][tok]
  float*  gT   = (float*) alloc((size_t)NTOK * CA * 4);   // [h*48+d][tok]
  float*  bias = (float*) alloc((size_t)NH * NTOK * NTOK * 4);
  bf16_t* ogb  = (bf16_t*)alloc((size_t)NTOK * CA * 2);
  (void)ws_size; (void)in_sizes; (void)n_in; (void)out_size;

  // 1) weight convert+transpose, pad-buffer init
  cvt_t_kernel<<<dim3(CS, CA / 256), 256, 0, stream>>>(gate_w, gwT, CS);
  cvt_t_kernel<<<dim3(CS, CA / 256), 256, 0, stream>>>(skip_w, swT, CS);
  cvt_t_kernel<<<dim3(CA, CA / 256), 256, 0, stream>>>(wq, wqT, CA);
  cvt_t_kernel<<<dim3(CA, CA / 256), 256, 0, stream>>>(wk, wkT, CA);
  cvt_t_kernel<<<dim3(CA, CA / 256), 256, 0, stream>>>(wv, wvT, CA);
  cvt_t_kernel<<<dim3(CA, CA / 256), 256, 0, stream>>>(wg, wgT, CA);
  cvt_t_kernel<<<dim3(CA, CA / 256), 256, 0, stream>>>(wo, woT, CA);
  cvt_t_kernel<<<dim3(CS, CA / 256), 256, 0, stream>>>(last_w, lwT, CS);
  int n4 = NH * NTOK * HDP * 2 / 16;
  zero4_kernel<<<(n4 + 255) / 256, 256, 0, stream>>>((uint4*)qpad, n4);
  zero4_kernel<<<(n4 + 255) / 256, 256, 0, stream>>>((uint4*)kpad, n4);

  // 2) layernorms
  ln_a_kernel<<<NTOK, 256, 0, stream>>>(a, anf);
  ln_s_kernel<<<NTOK, 128, 0, stream>>>(s, s_ln_w, snb, sbb);

  // 3) adaLN dual-GEMM -> a2
  adaln_gemm_kernel<<<dim3(NTOK / 16, CA / 64), 128, 0, stream>>>(snb, gwT, swT, gate_b, anf, a2b);

  // 4) QKVG quad-GEMM
  qkvg_kernel<<<dim3(NTOK / 16, CA / 64), 128, 0, stream>>>(a2b, wqT, wkT, wvT, wgT, bq, qpad, kpad, vT, gT);

  // 5) pair bias (single pass over 512MB z, WMMA projection)
  pair_bias_kernel<<<(NTOK * NTOK) / 32, 256, 0, stream>>>(z, z_ln_w, z_ln_b, wz, bias);

  // 6) QK^T + bias + softmax (probabilities written in place over bias)
  attn_softmax_kernel<<<dim3(NTOK / 16, NH), 256, 0, stream>>>(qpad, kpad, bias);

  // 7) P@V fused with sigmoid gate
  pv_kernel<<<dim3(NTOK / 64, NH), 128, 0, stream>>>(bias, vT, gT, ogb);

  // 8) output projection + final sigmoid gate
  out_gemm_kernel<<<dim3(NTOK / 16, CA / 64), 128, 0, stream>>>(ogb, woT, bo, sbb, lwT, last_b, out);
}
